// AttentionLayer_14817637171676
// MI455X (gfx1250) — compile-verified, matrix operand order checked by reading, not run on hardware
//
#include <hip/hip_runtime.h>
#include <hip/hip_bf16.h>

// ---------------- types ----------------
typedef __bf16 bf16_t;
typedef __attribute__((ext_vector_type(16))) __bf16 v16bf;
typedef __attribute__((ext_vector_type(8)))  __bf16 v8bf;
typedef __attribute__((ext_vector_type(8)))  float   v8f;
typedef __attribute__((ext_vector_type(4)))  float   v4f;

// ---------------- problem dims ----------------
#define BB   4
#define LL   8192
#define CC   1024
#define HH   8
#define DD   128
#define NTOK (BB*LL)          // 32768

// ---------------- fragment loaders ----------------
// bf16 A/B fragment: lane supplies K values {c..c+7, c+16..c+23} from a
// contiguous row (c = 8 for upper half-wave). Two b128 loads per fragment.
__device__ __forceinline__ v16bf load_frag_bf16(const bf16_t* __restrict__ rowp, int cbase) {
    v8bf a = *reinterpret_cast<const v8bf*>(rowp + cbase);
    v8bf b = *reinterpret_cast<const v8bf*>(rowp + cbase + 16);
    v16bf f;
#pragma unroll
    for (int j = 0; j < 8; ++j) { f[j] = a[j]; f[8 + j] = b[j]; }
    return f;
}

// fp32 source -> bf16 fragment (for the x operand of the QKV GEMM)
__device__ __forceinline__ v16bf load_frag_f32(const float* __restrict__ rowp, int cbase) {
    const v4f* p = reinterpret_cast<const v4f*>(rowp + cbase);
    v4f a0 = p[0], a1 = p[1];
    const v4f* q = reinterpret_cast<const v4f*>(rowp + cbase + 16);
    v4f a2 = q[0], a3 = q[1];
    v16bf f;
#pragma unroll
    for (int j = 0; j < 4; ++j) {
        f[j]      = (bf16_t)a0[j];
        f[4 + j]  = (bf16_t)a1[j];
        f[8 + j]  = (bf16_t)a2[j];
        f[12 + j] = (bf16_t)a3[j];
    }
    return f;
}

__device__ __forceinline__ float featmap(float f) {   // elu(x)+1
    return f > 0.0f ? f + 1.0f : __expf(f);
}

// ---------------- kernel A: weight convert + transpose to bf16 ----------------
// Wt[z][n][k] = (bf16) W_z[k][n]
__global__ __launch_bounds__(256) void wconv_kernel(const float* __restrict__ Wq,
                                                    const float* __restrict__ Wk,
                                                    const float* __restrict__ Wv,
                                                    bf16_t* __restrict__ Wt) {
    int idx = blockIdx.x * 256 + threadIdx.x;          // 3*C*C total
    int z   = idx / (CC * CC);
    int rem = idx % (CC * CC);
    int n   = rem / CC;
    int k   = rem % CC;
    const float* W = (z == 0) ? Wq : (z == 1) ? Wk : Wv;
    Wt[(size_t)z * CC * CC + (size_t)n * CC + k] = (bf16_t)W[(size_t)k * CC + n];
}

// ---------------- kernel B: fused QKV projection GEMM + feature map ----------------
// grid (NTOK/16, 1, 3); block 256 (8 waves). Block covers 16 token rows x 1024 cols;
// wave w covers cols [w*128, w*128+128) as 8 WMMA tiles sharing one A fragment.
__global__ __launch_bounds__(256) void qkv_kernel(const float* __restrict__ x,
                                                  const bf16_t* __restrict__ Wt,
                                                  const float* __restrict__ bq,
                                                  const float* __restrict__ bk,
                                                  const float* __restrict__ bv,
                                                  bf16_t* __restrict__ Qf,
                                                  bf16_t* __restrict__ Kt,
                                                  bf16_t* __restrict__ Vt) {
    const int z     = blockIdx.z;
    const int m0    = blockIdx.x * 16;                 // token row block
    const int wave  = threadIdx.x >> 5;
    const int lane  = threadIdx.x & 31;
    const int l16   = lane & 15;
    const int half8 = (lane >> 4) << 3;                // 0 or 8
    const int n0    = wave * 128;

    const bf16_t* Wz   = Wt + (size_t)z * CC * CC;
    const float*  bias = (z == 0) ? bq : (z == 1) ? bk : bv;
    const float*  xrow = x + (size_t)(m0 + l16) * CC;

    // per-tile weight row pointers (column n, contiguous over k thanks to transpose)
    const bf16_t* wrow[8];
#pragma unroll
    for (int t = 0; t < 8; ++t)
        wrow[t] = Wz + (size_t)(n0 + t * 16 + l16) * CC;

    v8f acc[8];
#pragma unroll
    for (int t = 0; t < 8; ++t) acc[t] = (v8f)(0.0f);

    for (int k0 = 0; k0 < CC; k0 += 32) {
        v16bf afrag = load_frag_f32(xrow, k0 + half8);
#pragma unroll
        for (int t = 0; t < 8; ++t) {
            v16bf bfrag = load_frag_bf16(wrow[t], k0 + half8);
            acc[t] = __builtin_amdgcn_wmma_f32_16x16x32_bf16(
                         false, afrag, false, bfrag, (short)0, acc[t], false, false);
        }
    }

    // epilogue: bias + activation + store.
    // Q: row-major [token][C]. K,V: transposed [bh][d][s] -- the 8 accumulator
    // rows per lane are CONSECUTIVE sequence positions there, so they pack
    // into one aligned 16B store per (lane, tile).
#pragma unroll
    for (int t = 0; t < 8; ++t) {
        int col = n0 + t * 16 + l16;
        float bb = bias[col];
        if (z == 0) {
#pragma unroll
            for (int r = 0; r < 8; ++r) {
                int g = m0 + r + half8;                // global token
                Qf[(size_t)g * CC + col] = (bf16_t)featmap(acc[t][r] + bb);
            }
        } else {
            v8bf pack;
#pragma unroll
            for (int r = 0; r < 8; ++r) {
                float f = acc[t][r] + bb;
                f = (z == 2) ? f * (1.0f / (float)LL) : featmap(f);
                pack[r] = (bf16_t)f;
            }
            int h = col >> 7, d = col & 127;
            int b = m0 >> 13;                          // 16-row block stays in one b
            int l = (m0 & (LL - 1)) + half8;           // first of 8 consecutive s
            bf16_t* dst = ((z == 2) ? Vt : Kt) +
                          (((size_t)(b * HH + h)) * DD + d) * LL + l;
            *reinterpret_cast<v8bf*>(dst) = pack;
        }
    }
}

// ---------------- kernel C: KV = K^T V per (b,h), contraction over s ----------------
// grid (8, 32): x = 16-row d1 block, y = bh. Wave w owns d2 tile w*16.
// Stores KVb transposed: KVb[bh][d2][d1] (bf16) for the final GEMM's B fragments.
__global__ __launch_bounds__(256) void kv_kernel(const bf16_t* __restrict__ Kt,
                                                 const bf16_t* __restrict__ Vt,
                                                 bf16_t* __restrict__ KVb) {
    const int bh    = blockIdx.y;
    const int wave  = threadIdx.x >> 5;
    const int lane  = threadIdx.x & 31;
    const int l16   = lane & 15;
    const int half8 = (lane >> 4) << 3;

    const bf16_t* arow = Kt + ((size_t)bh * DD + blockIdx.x * 16 + l16) * LL;
    const bf16_t* brow = Vt + ((size_t)bh * DD + wave * 16 + l16) * LL;

    v8f acc = (v8f)(0.0f);
    for (int s0 = 0; s0 < LL; s0 += 32) {
        v16bf af = load_frag_bf16(arow, s0 + half8);
        v16bf bf_ = load_frag_bf16(brow, s0 + half8);
        acc = __builtin_amdgcn_wmma_f32_16x16x32_bf16(
                  false, af, false, bf_, (short)0, acc, false, false);
    }

    // 8 accumulator rows per lane are consecutive d1 -> one packed 16B store
    int ncol  = wave * 16 + l16;                       // d2
    int mrow0 = blockIdx.x * 16 + half8;               // first of 8 consecutive d1
    v8bf pack;
#pragma unroll
    for (int r = 0; r < 8; ++r) pack[r] = (bf16_t)acc[r];
    *reinterpret_cast<v8bf*>(KVb + ((size_t)bh * DD + ncol) * DD + mrow0) = pack;
}

// ---------------- kernel D: Ksum[bh][d] = sum_s Kt[bh][d][s] ----------------
__global__ __launch_bounds__(256) void ksum_kernel(const bf16_t* __restrict__ Kt,
                                                   float* __restrict__ Ksum) {
    int row = blockIdx.x;                              // bh*D + d, 4096 rows
    const bf16_t* p = Kt + (size_t)row * LL;
    float s = 0.0f;
    for (int i = threadIdx.x * 8; i < LL; i += 256 * 8) {
        v8bf v = *reinterpret_cast<const v8bf*>(p + i);
#pragma unroll
        for (int e = 0; e < 8; ++e) s += (float)v[e];
    }
    __shared__ float red[256];
    red[threadIdx.x] = s;
    __syncthreads();
    for (int off = 128; off > 0; off >>= 1) {
        if ((int)threadIdx.x < off) red[threadIdx.x] += red[threadIdx.x + off];
        __syncthreads();
    }
    if (threadIdx.x == 0) Ksum[row] = red[0];
}

// ---------------- kernel E: Z[h][g] = 1 / (Q[g,h,:] . Ksum[bh,:] + eps) ----------------
// Z is laid out [h][token] so the output kernel reads 8 consecutive tokens
// as two v4f loads.
__global__ __launch_bounds__(256) void z_kernel(const bf16_t* __restrict__ Qf,
                                                const float* __restrict__ Ksum,
                                                float* __restrict__ Zbuf) {
    int idx = blockIdx.x * 256 + threadIdx.x;          // h*NTOK + g
    int h = idx / NTOK, g = idx % NTOK;
    int b = g >> 13, bh = b * HH + h;
    const bf16_t* qrow = Qf + (size_t)g * CC + h * DD;
    const float*  ks   = Ksum + (size_t)bh * DD;
    float dot = 0.0f;
    for (int j = 0; j < DD; j += 8) {
        v8bf qv = *reinterpret_cast<const v8bf*>(qrow + j);
#pragma unroll
        for (int e = 0; e < 8; ++e) dot += (float)qv[e] * ks[j + e];
    }
    Zbuf[idx] = 1.0f / (dot + 1e-6f);
}

// ---------------- kernel F: out = (Q @ KV) * Z * L ----------------
// grid (NTOK/16, H): block covers 16 tokens x one head (128 cols); wave w -> d2 tile w*16.
__global__ __launch_bounds__(256) void out_kernel(const bf16_t* __restrict__ Qf,
                                                  const bf16_t* __restrict__ KVb,
                                                  const float* __restrict__ Zbuf,
                                                  float* __restrict__ out) {
    const int m0    = blockIdx.x * 16;
    const int h     = blockIdx.y;
    const int wave  = threadIdx.x >> 5;
    const int lane  = threadIdx.x & 31;
    const int l16   = lane & 15;
    const int half8 = (lane >> 4) << 3;
    const int b     = m0 >> 13;
    const int bh    = b * HH + h;

    const bf16_t* arow = Qf + (size_t)(m0 + l16) * CC + h * DD;
    const bf16_t* brow = KVb + ((size_t)bh * DD + wave * 16 + l16) * DD;

    v8f acc = (v8f)(0.0f);
#pragma unroll
    for (int k0 = 0; k0 < DD; k0 += 32) {
        v16bf af = load_frag_bf16(arow, k0 + half8);
        v16bf bf_ = load_frag_bf16(brow, k0 + half8);
        acc = __builtin_amdgcn_wmma_f32_16x16x32_bf16(
                  false, af, false, bf_, (short)0, acc, false, false);
    }

    // Z for this lane's 8 consecutive tokens: two contiguous v4f loads
    const float* zp = Zbuf + (size_t)h * NTOK + (m0 + half8);
    v4f z0 = *reinterpret_cast<const v4f*>(zp);
    v4f z1 = *reinterpret_cast<const v4f*>(zp + 4);

    int ncol = wave * 16 + l16;                        // d2 within head
#pragma unroll
    for (int r = 0; r < 8; ++r) {
        int g = m0 + r + half8;
        float zz = (r < 4) ? z0[r & 3] : z1[r & 3];
        out[(size_t)g * CC + h * DD + ncol] = acc[r] * zz * (float)LL;
    }
}

// ---------------- launch ----------------
extern "C" void kernel_launch(void* const* d_in, const int* in_sizes, int n_in,
                              void* d_out, int out_size, void* d_ws, size_t ws_size,
                              hipStream_t stream) {
    const float* x  = (const float*)d_in[0];
    const float* Wq = (const float*)d_in[1];
    const float* bq = (const float*)d_in[2];
    const float* Wk = (const float*)d_in[3];
    const float* bk = (const float*)d_in[4];
    const float* Wv = (const float*)d_in[5];
    const float* bv = (const float*)d_in[6];
    float* out = (float*)d_out;

    // workspace carving (all offsets 256B aligned)
    char* ws = (char*)d_ws;
    size_t off = 0;
    bf16_t* Wt  = (bf16_t*)(ws + off); off += (size_t)3 * CC * CC * sizeof(bf16_t);   // 6 MB
    bf16_t* Qf  = (bf16_t*)(ws + off); off += (size_t)NTOK * CC * sizeof(bf16_t);     // 64 MB
    bf16_t* Kt  = (bf16_t*)(ws + off); off += (size_t)NTOK * CC * sizeof(bf16_t);     // 64 MB
    bf16_t* Vt  = (bf16_t*)(ws + off); off += (size_t)NTOK * CC * sizeof(bf16_t);     // 64 MB
    bf16_t* KVb = (bf16_t*)(ws + off); off += (size_t)BB * HH * DD * DD * sizeof(bf16_t); // 1 MB
    float*  Ksum= (float*)(ws + off);  off += (size_t)BB * HH * DD * sizeof(float);   // 16 KB
    float*  Zb  = (float*)(ws + off);  off += (size_t)NTOK * HH * sizeof(float);      // 1 MB

    // A: weights -> bf16 transposed
    wconv_kernel<<<(3 * CC * CC) / 256, 256, 0, stream>>>(Wq, Wk, Wv, Wt);
    // B: fused QKV projection + feature map (+ K/V transpose)
    qkv_kernel<<<dim3(NTOK / 16, 1, 3), 256, 0, stream>>>(x, Wt, bq, bk, bv, Qf, Kt, Vt);
    // D: Ksum
    ksum_kernel<<<BB * HH * DD, 256, 0, stream>>>(Kt, Ksum);
    // C: KV outer-product accumulation over sequence
    kv_kernel<<<dim3(DD / 16, BB * HH), 256, 0, stream>>>(Kt, Vt, KVb);
    // E: per-(token,head) normalizer
    z_kernel<<<(NTOK * HH) / 256, 256, 0, stream>>>(Qf, Ksum, Zb);
    // F: final GEMM + normalize
    out_kernel<<<dim3(NTOK / 16, HH), 256, 0, stream>>>(Qf, KVb, Zb, out);
}
